// KANLayer_66657892434504
// MI455X (gfx1250) — matrix operand order repaired and checked by reading, not compiled, not run on hardware
//
#include <hip/hip_runtime.h>
#include <stdint.h>

#define IN_F 256
#define OUT_F 64
#define KPF 16                    // padded K slots per input feature (silu + 13 basis + 2 zero)
#define CI 16                     // input features per macro-iteration
#define KCH (CI * KPF)            // 256 K per macro-iteration -> 8 WMMA k-steps
#define NMACRO (IN_F / CI)        // 16
#define WAVE_M 2                  // 16-row M tiles per wave
#define ROWS_PER_WAVE (16 * WAVE_M)              // 32
#define WAVES 4
#define ROWS_PER_BLOCK (ROWS_PER_WAVE * WAVES)   // 128
#define NTILES (OUT_F / 16)       // 4
#define KSTEPS (KCH / 32)         // 8

typedef __attribute__((ext_vector_type(16))) __bf16 v16bf;
typedef __attribute__((ext_vector_type(8)))  float  v8f;
typedef __attribute__((ext_vector_type(4)))  unsigned int u32x4;
typedef __attribute__((ext_vector_type(8)))  int i32x8;
typedef __attribute__((ext_vector_type(4)))  int i32x4;

union FragU { v16bf v; uint4 q[2]; };

// round-to-nearest-even fp32 -> bf16 bit pattern
__device__ __forceinline__ unsigned short f2bf(float f) {
    union { float f; unsigned u; } v; v.f = f;
    unsigned u = v.u;
    unsigned r = (u + 0x7FFFu + ((u >> 16) & 1u)) >> 16;
    return (unsigned short)r;
}

// ---- prep: interleave + pad + convert weights to bf16: Wg[o][i*16+s], row stride 4096 ----
__global__ __launch_bounds__(256) void kan_prep_weights(
    const float* __restrict__ wbase, const float* __restrict__ wspl,
    unsigned short* __restrict__ wg)
{
    int p = blockIdx.x * 256 + threadIdx.x;  // 0..16383 = (o,i) pairs
    int o = p >> 8;
    int i = p & 255;
    __align__(16) unsigned short wv[16];
    wv[0] = f2bf(wbase[o * IN_F + i]);
    const float* sp = wspl + ((size_t)o * IN_F + i) * 13;
    #pragma unroll
    for (int s = 0; s < 13; ++s) wv[1 + s] = f2bf(sp[s]);
    wv[14] = 0; wv[15] = 0;
    uint4* dst = (uint4*)&wg[((size_t)o * IN_F + i) * KPF];
    dst[0] = *(const uint4*)&wv[0];
    dst[1] = *(const uint4*)&wv[8];
}

template <bool USE_WS>
__global__ __launch_bounds__(128) void kan_main(
    const float* __restrict__ x,              // (B, 256)
    const float* __restrict__ wbase,          // (64, 256)      [fallback path]
    const float* __restrict__ wspl,           // (64, 256, 13)  [fallback path]
    const float* __restrict__ grid,           // (17,)
    const unsigned short* __restrict__ wg,    // (64, 4096) bf16 [TDM path]
    float* __restrict__ out)                  // (B, 64)
{
    __shared__ unsigned short Alds[WAVES][ROWS_PER_WAVE][KCH];  // 64 KB
    __shared__ unsigned short Wlds[OUT_F][KCH];                 // 32 KB

    const int tid  = threadIdx.x;
    const int lane = tid & 31;
    const int wave = tid >> 5;
    const int row0 = blockIdx.x * ROWS_PER_BLOCK;

    // uniform grid -> scalar loads; closed-form cubic B-spline uses only g0, h
    const float g0v  = grid[0];
    const float h    = grid[1] - g0v;
    const float invh = 1.0f / h;

    v8f acc[WAVE_M][NTILES];
    #pragma unroll
    for (int mt = 0; mt < WAVE_M; ++mt)
        #pragma unroll
        for (int t = 0; t < NTILES; ++t) {
            v8f z = {0.f, 0.f, 0.f, 0.f, 0.f, 0.f, 0.f, 0.f};
            acc[mt][t] = z;
        }

    for (int mi = 0; mi < NMACRO; ++mi) {
        const int i0 = mi * CI;
        __syncthreads();   // previous iteration's LDS readers are done

        if (USE_WS) {
            // ---- TDM: DMA 64 x 256 bf16 weight slice (row stride 4096) into Wlds ----
            if (wave == 0) {
                unsigned long long ga = (unsigned long long)(uintptr_t)(wg + (size_t)i0 * KPF);
                unsigned ldsoff = (unsigned)(uintptr_t)(&Wlds[0][0]);
                u32x4 g0;
                g0[0] = 1u;                                  // count=1 (valid descriptor)
                g0[1] = ldsoff;                              // lds_addr (bytes)
                g0[2] = (unsigned)(ga & 0xFFFFFFFFu);        // global_addr[31:0]
                g0[3] = (unsigned)((ga >> 32) & 0x1FFFFFFu)  // global_addr[56:32]
                      | (2u << 30);                          // type=2 ("image")
                i32x8 g1;
                g1[0] = (int)(1u << 16);                     // data_size=1 (2 bytes)
                g1[1] = (int)((4096u & 0xFFFFu) << 16);      // tensor_dim0 lo16 -> bits 63:48
                g1[2] = (int)(((4096u >> 16) & 0xFFFFu)      // tensor_dim0 hi16
                      | (64u << 16));                        // tensor_dim1 lo16
                g1[3] = (int)(256u << 16);                   // tensor_dim1 hi16=0 | tile_dim0=256
                g1[4] = (int)64u;                            // tile_dim1=64, tile_dim2=0
                g1[5] = (int)4096u;                          // tensor_dim0_stride lo32
                g1[6] = 0;                                   // stride hi16 | dim1_stride lo16
                g1[7] = 0;
                i32x4 z4 = {0, 0, 0, 0};
                i32x8 z8 = {0, 0, 0, 0, 0, 0, 0, 0};
                __builtin_amdgcn_tensor_load_to_lds(g0, g1, z4, z4, z8, 0);
            }
        } else {
            // ---- fallback: convert-stage weights (8 (o,i) pairs per thread) ----
            #pragma unroll
            for (int e = 0; e < 8; ++e) {
                int p  = tid + 128 * e;
                int o  = p >> 4;
                int il = p & 15;
                int i  = i0 + il;
                __align__(16) unsigned short wv[16];
                wv[0] = f2bf(wbase[o * IN_F + i]);
                const float* sp = wspl + ((size_t)o * IN_F + i) * 13;
                #pragma unroll
                for (int s = 0; s < 13; ++s) wv[1 + s] = f2bf(sp[s]);
                wv[14] = 0; wv[15] = 0;
                uint4* dst = (uint4*)&Wlds[o][il * KPF];
                dst[0] = *(const uint4*)&wv[0];
                dst[1] = *(const uint4*)&wv[8];
            }
        }

        // ---- zero-fill this wave's A region (16 KB): 32 b128 stores per lane ----
        {
            uint4 zz = {0, 0, 0, 0};
            uint4* az = (uint4*)&Alds[wave][0][0];
            #pragma unroll
            for (int z = 0; z < 32; ++z) az[z * 32 + lane] = zz;
        }

        // ---- activation expansion: closed-form uniform cubic B-spline + silu ----
        // (overlaps with the TDM weight DMA)
        for (int e = 0; e < 16; ++e) {
            int eid = e * 32 + lane;       // 0..511
            int il  = eid & 15;            // feature within slice (coalesced)
            int r   = eid >> 4;            // row 0..31
            float xv = x[(size_t)(row0 + wave * ROWS_PER_WAVE + r) * IN_F + i0 + il];

            float sig  = 1.0f / (1.0f + __expf(-xv));
            float silu = xv * sig;
            float xc   = fminf(fmaxf(xv, -2.5f), 2.5f);

            float u   = (xc - g0v) * invh;          // in [1.75, 14.25]
            int   idx = (int)floorf(u);             // knot interval
            float t   = u - (float)idx;
            float omt = 1.0f - t;
            float t2 = t * t, t3 = t2 * t;
            const float c6 = 1.0f / 6.0f;
            float q[4];
            q[0] = omt * omt * omt * c6;
            q[1] = (3.0f * t3 - 6.0f * t2 + 4.0f) * c6;
            q[2] = (-3.0f * t3 + 3.0f * t2 + 3.0f * t + 1.0f) * c6;
            q[3] = t3 * c6;

            unsigned short* arow = &Alds[wave][r][il * KPF];
            arow[0] = f2bf(silu);
            int jb = idx - 3;
            #pragma unroll
            for (int qq = 0; qq < 4; ++qq) {
                int j = jb + qq;                    // basis index 0..12 kept (as reference)
                if (j >= 0 && j <= 12) arow[1 + j] = f2bf(q[qq]);
            }
        }

        if (USE_WS) {
            if (wave == 0) __builtin_amdgcn_s_wait_tensorcnt(0);
        }
        __syncthreads();

        // ---- WMMA: per k-step load 2 A + 4 B fragments, then 8 back-to-back WMMAs ----
        #pragma unroll
        for (int ks = 0; ks < KSTEPS; ++ks) {
            FragU a[WAVE_M];
            const int abase = ks * 32 + ((lane < 16) ? 0 : 8);
            #pragma unroll
            for (int mt = 0; mt < WAVE_M; ++mt) {
                const unsigned short* arow = &Alds[wave][mt * 16 + (lane & 15)][0];
                a[mt].q[0] = *(const uint4*)&arow[abase];
                a[mt].q[1] = *(const uint4*)&arow[abase + 16];
            }
            FragU bb[NTILES];
            const int bbase = ks * 32 + ((lane < 16) ? 0 : 8);
            #pragma unroll
            for (int t = 0; t < NTILES; ++t) {
                const unsigned short* brow = &Wlds[t * 16 + (lane & 15)][0];
                bb[t].q[0] = *(const uint4*)&brow[bbase];
                bb[t].q[1] = *(const uint4*)&brow[bbase + 16];
            }
            #pragma unroll
            for (int mt = 0; mt < WAVE_M; ++mt)
                #pragma unroll
                for (int t = 0; t < NTILES; ++t)
                    acc[mt][t] = __builtin_amdgcn_wmma_f32_16x16x32_bf16(
                        false, a[mt].v, false, bb[t].v, (short)0, acc[mt][t], false, false);
        }
    }

    // ---- store: f32 C/D layout (VGPR j -> M=j lanes 0-15, M=j+8 lanes 16-31) ----
    const int col  = lane & 15;
    const int rofs = (lane < 16) ? 0 : 8;
    #pragma unroll
    for (int mt = 0; mt < WAVE_M; ++mt) {
        const int rbase = row0 + wave * ROWS_PER_WAVE + mt * 16 + rofs;
        #pragma unroll
        for (int t = 0; t < NTILES; ++t)
            #pragma unroll
            for (int j = 0; j < 8; ++j)
                out[(size_t)(rbase + j) * OUT_F + t * 16 + col] = acc[mt][t][j];
    }
}

extern "C" void kernel_launch(void* const* d_in, const int* in_sizes, int n_in,
                              void* d_out, int out_size, void* d_ws, size_t ws_size,
                              hipStream_t stream) {
    const float* x     = (const float*)d_in[0];
    const float* wbase = (const float*)d_in[1];
    const float* wspl  = (const float*)d_in[2];
    const float* grid  = (const float*)d_in[3];
    float* out = (float*)d_out;

    int B = in_sizes[0] / IN_F;                  // 65536
    int blocks = B / ROWS_PER_BLOCK;             // 512

    size_t need = (size_t)OUT_F * IN_F * KPF * sizeof(unsigned short);  // 512 KB
    if (ws_size >= need) {
        unsigned short* wg = (unsigned short*)d_ws;
        kan_prep_weights<<<(OUT_F * IN_F) / 256, 256, 0, stream>>>(wbase, wspl, wg);
        kan_main<true><<<blocks, 128, 0, stream>>>(x, wbase, wspl, grid, wg, out);
    } else {
        kan_main<false><<<blocks, 128, 0, stream>>>(x, wbase, wspl, grid, nullptr, out);
    }
}
